// DeeperGCN_62199716381242
// MI455X (gfx1250) — matrix operand order repaired and checked by reading, compiled, not run on hardware
//
#include <hip/hip_runtime.h>

#define H      256
#define NN     20000
#define EE     320000
#define LL     7
#define GG     512
#define OUTD   128
#define ATOMV  128
#define BONDV  8
#define MEPS   1e-7f
#define BN_EPS 1e-5f

typedef __attribute__((ext_vector_type(16))) _Float16 v16h;
typedef __attribute__((ext_vector_type(8)))  _Float16 v8h;
typedef __attribute__((ext_vector_type(8)))  float    v8f;

// LDS byte-offset of a __shared__ object (addrspace(3) pointer value == LDS offset)
__device__ __forceinline__ unsigned lds_off(const void* p) {
  return (unsigned)(unsigned long long)(__attribute__((address_space(3))) const void*)p;
}

// ---------------- Atom encoder: hv[i,h] = sum_c atom_emb[c, nf[i,c], h] ----
__global__ __launch_bounds__(256) void k_atom_encode(
    const int* __restrict__ nf, const float* __restrict__ emb,
    float* __restrict__ hv) {
  const int i = blockIdx.x;
  const int h = threadIdx.x;
  float s = 0.f;
#pragma unroll
  for (int c = 0; c < 9; ++c) {
    const int v = nf[i * 9 + c];
    s += emb[((size_t)c * ATOMV + v) * H + h];
  }
  hv[(size_t)i * H + h] = s;
}

// ---------------- Generic zero fill --------------------------------------
__global__ void k_zero(float* __restrict__ p, long n) {
  long i = blockIdx.x * (long)blockDim.x + threadIdx.x;
  const long stride = (long)gridDim.x * blockDim.x;
  for (; i < n; i += stride) p[i] = 0.f;
}

// ---------------- BN batch stats: partial sums -> atomics (L2) -----------
__global__ __launch_bounds__(256) void k_bn_stats(
    const float* __restrict__ hv, float* __restrict__ sum,
    float* __restrict__ sq) {
  const int h = threadIdx.x;
  const int i0 = blockIdx.x * 64;
  const int i1 = (i0 + 64 < NN) ? i0 + 64 : NN;
  float s = 0.f, ss = 0.f;
  for (int i = i0; i < i1; ++i) {
    const float v = hv[(size_t)i * H + h];
    s += v;
    ss += v * v;
  }
  atomicAdd(&sum[h], s);
  atomicAdd(&sq[h], ss);
}

// ---------------- BN normalize + ReLU -> h1 -------------------------------
__global__ __launch_bounds__(256) void k_bn_apply(
    const float* __restrict__ hv, const float* __restrict__ sum,
    const float* __restrict__ sq, const float* __restrict__ gamma,
    const float* __restrict__ beta, float* __restrict__ h1) {
  const int i = blockIdx.x;
  const int h = threadIdx.x;
  const float mu  = sum[h] * (1.f / NN);
  const float var = sq[h] * (1.f / NN) - mu * mu;
  const size_t idx = (size_t)i * H + h;
  const float x = (hv[idx] - mu) * rsqrtf(var + BN_EPS) * gamma[h] + beta[h];
  h1[idx] = fmaxf(x, 0.f);
}

// ---------------- Edge message (recomputed per pass; h1 is L2-resident) ---
__device__ __forceinline__ float edge_msg(const float* __restrict__ h1,
                                          const float* __restrict__ bemb,
                                          const int* __restrict__ ef, int e,
                                          int s, int h) {
  float he = 0.f;
#pragma unroll
  for (int c = 0; c < 3; ++c) {
    const int f = ef[e * 3 + c];
    he += bemb[((size_t)c * BONDV + f) * H + h];
  }
  const float m = h1[(size_t)s * H + h] + he;
  return fmaxf(m, 0.f) + MEPS;  // m > 0 always (int-bit atomicMax is safe)
}

// Pass 1: segment max over incoming edges (float via int bits; m > 0)
__global__ __launch_bounds__(256) void k_edge_max(
    const float* __restrict__ h1, const float* __restrict__ bemb,
    const int* __restrict__ ef, const int* __restrict__ src,
    const int* __restrict__ dst, int* __restrict__ mmax) {
  const int e = blockIdx.x;
  const int h = threadIdx.x;
  const int s = src[e], d = dst[e];
  const float m = edge_msg(h1, bemb, ef, e, s, h);
  atomicMax(&mmax[(size_t)d * H + h], __float_as_int(m));
}

// Pass 2: denom = sum exp(m-max); num = sum m*exp(m-max)
__global__ __launch_bounds__(256) void k_edge_soft(
    const float* __restrict__ h1, const float* __restrict__ bemb,
    const int* __restrict__ ef, const int* __restrict__ src,
    const int* __restrict__ dst, const int* __restrict__ mmax,
    float* __restrict__ denom, float* __restrict__ num) {
  const int e = blockIdx.x;
  const int h = threadIdx.x;
  const int s = src[e], d = dst[e];
  const float m  = edge_msg(h1, bemb, ef, e, s, h);
  const size_t idx = (size_t)d * H + h;
  const float mx = __int_as_float(mmax[idx]);
  const float ex = __expf(m - mx);
  atomicAdd(&denom[idx], ex);
  atomicAdd(&num[idx], m * ex);
}

// ---------------- X = h1 + agg, converted to f16 for WMMA -----------------
__global__ __launch_bounds__(256) void k_xcompute(
    const float* __restrict__ h1, const float* __restrict__ denom,
    const float* __restrict__ num, _Float16* __restrict__ Xh) {
  const size_t idx = (size_t)blockIdx.x * H + threadIdx.x;
  const float dn = denom[idx];
  const float agg = dn > 0.f ? num[idx] / dn : 0.f;
  Xh[idx] = (_Float16)(h1[idx] + agg);
}

// ---------------- W[k,n] f32 -> Wt[n,k] f16 (128 KB, L2-resident) ---------
__global__ __launch_bounds__(256) void k_wconv(const float* __restrict__ W,
                                               _Float16* __restrict__ Wt) {
  const int k = blockIdx.x;
  const int n = threadIdx.x;
  Wt[(size_t)n * H + k] = (_Float16)W[(size_t)k * H + n];
}

// ---------------- WMMA GEMM: hv += Xh @ W + b (f16 in, f32 acc) -----------
// Block = 4 waves (128 thr), covers 16 rows x 256 cols.
// A tile (16x256 f16 = 8 KB, contiguous rows of Xh, shared by all 4 waves)
// is staged once into LDS via global_load_async_to_lds_b128 (ASYNCcnt path),
// then read as ds_load_b128 fragments. B is wave-private -> direct L2 loads.
__global__ __launch_bounds__(128) void k_gemm_wmma(
    const _Float16* __restrict__ Xh, const _Float16* __restrict__ Wt,
    const float* __restrict__ bias, float* __restrict__ hv) {
  __shared__ _Float16 As[16 * H];  // 8 KB

  const int t    = threadIdx.x;
  const int lane = t & 31;
  const int wave = t >> 5;
  const int row0 = blockIdx.x * 16;      // N = 20000 = 1250 * 16, exact
  const int col0 = wave * 64;

  // ---- async stage: 8192 B = 512 x b128; 128 threads x 4 chunks ----
  {
    const _Float16* gsrc = Xh + (size_t)row0 * H;
    const unsigned lbase = lds_off(As);
#pragma unroll
    for (int it = 0; it < 4; ++it) {
      const int chunk = it * 128 + t;                 // 16-byte chunk id
      const unsigned loff = lbase + chunk * 16;       // LDS byte offset
      const _Float16* ga = gsrc + chunk * 8;          // 8 halfs = 16 B
      asm volatile("global_load_async_to_lds_b128 %0, %1, off"
                   :: "v"(loff), "v"(ga) : "memory");
    }
    asm volatile("s_wait_asynccnt 0" ::: "memory");
    __syncthreads();  // make the tile visible to all 4 waves
  }

  v8f acc[4] = {};

  // A fragment addressing (contiguous k order, k = 16*(lane>>4) + s, matched by B):
  //   lanes 0-15: M=lane,    K in {0..15}
  //   lanes 16-31: M=lane-16, K in {16..31}
  const _Float16* arow = As + (size_t)(lane & 15) * H + ((lane >> 4) * 16);
  // B fragment: lanes 0-15 hold K=0..15 of col n=lane; lanes 16-31 K=16..31
  const int bk = (lane >> 4) * 16;

  for (int kt = 0; kt < 8; ++kt) {
    const int kb = kt * 32;
    union { v16h v; v8h h[2]; } a;
    a.h[0] = *(const v8h*)(arow + kb);        // ds_load_b128
    a.h[1] = *(const v8h*)(arow + kb + 8);    // ds_load_b128
#pragma unroll
    for (int j = 0; j < 4; ++j) {
      const _Float16* bp =
          Wt + (size_t)(col0 + j * 16 + (lane & 15)) * H + kb + bk;
      const v16h b = *(const v16h*)bp;        // 32 B global load (L2)
      acc[j] = __builtin_amdgcn_wmma_f32_16x16x32_f16(
          false, a.v, false, b, (short)0, acc[j], false, false);
    }
  }

  // Epilogue: C/D layout -> row = row0 + v + 8*(lane>>4), col = lane&15
  // hv_new = acc + bias + hv_old (residual), in place (block-private rows).
#pragma unroll
  for (int j = 0; j < 4; ++j) {
    const int col = col0 + j * 16 + (lane & 15);
    const int rbase = row0 + ((lane >> 4) * 8);
    const float bc = bias[col];
#pragma unroll
    for (int v = 0; v < 8; ++v) {
      const size_t idx = (size_t)(rbase + v) * H + col;
      hv[idx] = hv[idx] + acc[j][v] + bc;
    }
  }
}

// ---------------- Graph mean pooling --------------------------------------
__global__ __launch_bounds__(256) void k_pool(const float* __restrict__ hv,
                                              const int* __restrict__ gid,
                                              float* __restrict__ hg,
                                              float* __restrict__ cnt) {
  const int i = blockIdx.x;
  const int h = threadIdx.x;
  const int g = gid[i];
  atomicAdd(&hg[(size_t)g * H + h], hv[(size_t)i * H + h]);
  if (h == 0) atomicAdd(&cnt[g], 1.f);
}

// ---------------- Output head: (hg/cnt) @ w_out + b_out -------------------
__global__ __launch_bounds__(128) void k_out(const float* __restrict__ hg,
                                             const float* __restrict__ cnt,
                                             const float* __restrict__ wo,
                                             const float* __restrict__ bo,
                                             float* __restrict__ out) {
  const int g = blockIdx.x;
  const int o = threadIdx.x;
  float acc = 0.f;
  for (int k = 0; k < H; ++k)
    acc += hg[(size_t)g * H + k] * wo[(size_t)k * OUTD + o];
  const float c = fmaxf(cnt[g], 1.f);
  out[(size_t)g * OUTD + o] = acc / c + bo[o];
}

// ==========================================================================
extern "C" void kernel_launch(void* const* d_in, const int* in_sizes, int n_in,
                              void* d_out, int out_size, void* d_ws,
                              size_t ws_size, hipStream_t stream) {
  (void)in_sizes; (void)n_in; (void)out_size; (void)ws_size;
  const int*   node_feats = (const int*)d_in[0];
  const int*   edge_feats = (const int*)d_in[1];
  const int*   src        = (const int*)d_in[2];
  const int*   dst        = (const int*)d_in[3];
  const int*   gid        = (const int*)d_in[4];
  const float* atom_emb   = (const float*)d_in[5];
  const float* bond_emb   = (const float*)d_in[6];
  const float* bn_gamma   = (const float*)d_in[7];
  const float* bn_beta    = (const float*)d_in[8];
  const float* mlp_w      = (const float*)d_in[9];
  const float* mlp_b      = (const float*)d_in[10];
  const float* w_out      = (const float*)d_in[11];
  const float* b_out      = (const float*)d_in[12];
  float* out = (float*)d_out;

  char* ws = (char*)d_ws;
  size_t off = 0;
  auto alloc = [&](size_t bytes) -> void* {
    off = (off + 255) & ~(size_t)255;
    void* p = ws + off;
    off += bytes;
    return p;
  };

  float*    hv = (float*)alloc((size_t)NN * H * 4);
  float*    h1 = (float*)alloc((size_t)NN * H * 4);
  _Float16* Xh = (_Float16*)alloc((size_t)NN * H * 2);
  _Float16* Wt = (_Float16*)alloc((size_t)H * H * 2);
  // hg + cnt contiguous (single zero fill)
  float* hg  = (float*)alloc(((size_t)GG * H + GG) * 4);
  float* cnt = hg + (size_t)GG * H;
  // sum, sq, mmax, denom, num contiguous (single zero fill per layer)
  const size_t zcount = 2 * H + 3 * (size_t)NN * H;
  float* zblock = (float*)alloc(zcount * 4);
  float* sum   = zblock;
  float* sq    = zblock + H;
  int*   mmax  = (int*)(zblock + 2 * H);
  float* denom = zblock + 2 * H + (size_t)NN * H;
  float* num   = zblock + 2 * H + 2 * (size_t)NN * H;

  k_atom_encode<<<NN, H, 0, stream>>>(node_feats, atom_emb, hv);

  for (int l = 0; l < LL; ++l) {
    k_zero<<<1024, 256, 0, stream>>>(zblock, (long)zcount);
    k_bn_stats<<<(NN + 63) / 64, H, 0, stream>>>(hv, sum, sq);
    k_bn_apply<<<NN, H, 0, stream>>>(hv, sum, sq, bn_gamma + l * H,
                                     bn_beta + l * H, h1);
    k_wconv<<<H, H, 0, stream>>>(mlp_w + (size_t)l * H * H, Wt);
    const float* bemb = bond_emb + (size_t)l * 3 * BONDV * H;
    k_edge_max<<<EE, H, 0, stream>>>(h1, bemb, edge_feats, src, dst, mmax);
    k_edge_soft<<<EE, H, 0, stream>>>(h1, bemb, edge_feats, src, dst, mmax,
                                      denom, num);
    k_xcompute<<<NN, H, 0, stream>>>(h1, denom, num, Xh);
    k_gemm_wmma<<<NN / 16, 128, 0, stream>>>(Xh, Wt, mlp_b + l * H, hv);
  }

  k_zero<<<64, 256, 0, stream>>>(hg, (long)((size_t)GG * H + GG));
  k_pool<<<NN, H, 0, stream>>>(hv, gid, hg, cnt);
  k_out<<<GG, OUTD, 0, stream>>>(hg, cnt, w_out, b_out, out);
}
